// EncodecFeatures_91087666413730
// MI455X (gfx1250) — matrix operand order repaired and checked by reading, not compiled, not run on hardware
//
#include <hip/hip_runtime.h>

// ---------------------------------------------------------------------------
// EnCodec encoder + RVQ for MI455X (gfx1250, wave32, WMMA + TDM).
// - Activations bf16 (HBM-bound: halve traffic, feed v_wmma_f32_16x16x32_bf16)
// - Weight tiles DMA'd to LDS by the Tensor Data Mover (tensor_load_to_lds,
//   s_wait_tensorcnt), overlapped with VALU im2col+ELU staging.
// - RVQ distance GEMM in full f32 via v_wmma_f32_16x16x4_f32 (argmin fidelity)
// ---------------------------------------------------------------------------

typedef __attribute__((ext_vector_type(16))) __bf16       v16bf;
typedef __attribute__((ext_vector_type(8)))  float        v8f;
typedef __attribute__((ext_vector_type(2)))  float        v2f;
typedef __attribute__((ext_vector_type(4)))  unsigned int v4u;
typedef __attribute__((ext_vector_type(8)))  int          v8i;
typedef __attribute__((ext_vector_type(4)))  int          v4i;

static __device__ __forceinline__ float elu_f(float x) {
  return x > 0.f ? x : (__expf(x) - 1.f);
}

// ---------------- weight f32 -> bf16 with [co][ci][k] -> [co][k][ci] permute
__global__ void k_w2bf(const float* __restrict__ w, __bf16* __restrict__ wt,
                       int cout, int cin, int k) {
  int idx = blockIdx.x * blockDim.x + threadIdx.x;
  int n = cout * cin * k;
  if (idx >= n) return;
  int co  = idx / (cin * k);
  int rem = idx % (cin * k);
  int ci  = rem / k;
  int dk  = rem % k;
  wt[(size_t)co * cin * k + dk * cin + ci] = (__bf16)w[idx];
}

// ---------------- first conv: 1 -> 32 channels, k=7 causal (tiny, scalar)
__global__ void k_conv_in(const float* __restrict__ x, const float* __restrict__ w,
                          const float* __restrict__ b, __bf16* __restrict__ out) {
  long idx = (long)blockIdx.x * blockDim.x + threadIdx.x;
  const long total = (long)16 * 32 * 240000;
  if (idx >= total) return;
  int t  = (int)(idx % 240000);
  int c  = (int)((idx / 240000) % 32);
  int bb = (int)(idx / (240000L * 32));
  float acc = b[c];
#pragma unroll
  for (int j = 0; j < 7; ++j) {
    int ti = t + j - 6;                      // left pad = k - stride = 6
    if (ti >= 0) acc += w[c * 7 + j] * x[(long)bb * 240000 + ti];
  }
  out[idx] = (__bf16)acc;
}

// ---------------- generic causal conv as WMMA GEMM -------------------------
// D[m=c_out, n=time] = W[m,K] * im2col(elu(in))[K, n],  K = CIN*KSIZE
// MODE 0: write bf16   MODE 1: residual add in place   MODE 2: f32 r-layout
template <int CIN, int KSIZE, int STRIDE, int MODE>
__global__ __launch_bounds__(128)
void k_conv_gemm(const __bf16* __restrict__ in, const __bf16* __restrict__ wT,
                 const float* __restrict__ bias,
                 __bf16* __restrict__ out_bf, float* __restrict__ out_f32,
                 int cout, int Tin, int Tout) {
  constexpr int K    = CIN * KSIZE;
  constexpr int Kpad = (K + 31) & ~31;
  constexpr int PAD  = KSIZE - STRIDE;

  __shared__ __bf16 Bs[32][18];            // K x N activation tile (+pad)
  __shared__ __bf16 As[4][16][34];         // TDM-filled, 68B row pitch (pad)

  const int tid  = threadIdx.x;
  const int lane = tid & 31;
  const int wave = tid >> 5;
  const int lm   = lane & 15;
  const bool hi  = lane >= 16;
  const int t0   = blockIdx.x * 16;
  const int MT   = cout >> 4;
  const int mt   = blockIdx.y * 4 + wave;
  const int m0   = mt * 16;
  const int bb   = blockIdx.z;
  const __bf16* inb = in + (size_t)bb * CIN * Tin;

  const unsigned lds_off =
      (unsigned)(uintptr_t)(void*)&As[wave][0][0];   // low 32b = LDS byte addr

  v8f acc = {};
  for (int kk0 = 0; kk0 < Kpad; kk0 += 32) {
    // --- kick TDM: DMA this wave's 16x32 weight tile into its LDS region.
    //     tensor_dim0 = K-kk0 -> OOB zero-fill covers the K tail;
    //     pad_interval=16 DWORDs + pad_amount=1 DWORD -> 68B LDS row pitch.
    if (mt < MT) {
      unsigned long long ga =
          (unsigned long long)(uintptr_t)(wT + (size_t)m0 * K + kk0);
      v4u g0;
      g0[0] = 1u;                                    // count=1, user mode
      g0[1] = lds_off;                               // lds_addr
      g0[2] = (unsigned)ga;                          // global_addr[31:0]
      g0[3] = (unsigned)((ga >> 32) & 0x01ffffffu) | (2u << 30);  // addr|type=2
      unsigned td0 = (unsigned)(K - kk0);
      v8i g1;
      g1[0] = (int)0x00D10000u;       // data_size=2B, pad_en, pad_interval=3
      g1[1] = (int)((td0 & 0xffffu) << 16);          // tensor_dim0 lo
      g1[2] = (int)((td0 >> 16) | (16u << 16));      // dim0 hi | tensor_dim1=16
      g1[3] = (int)(32u << 16);                      // tile_dim0 = 32 elems
      g1[4] = (int)16u;                              // tile_dim1 = 16 rows
      g1[5] = (int)(unsigned)K;                      // tensor_dim0_stride
      g1[6] = 0;
      g1[7] = 0;
      v4i gz4 = {0, 0, 0, 0};                        // 2-D tensor: groups 2/3
      v8i gz8 = {0, 0, 0, 0, 0, 0, 0, 0};
      __builtin_amdgcn_tensor_load_to_lds(g0, g1, gz4, gz4, gz8, 0);
    }
    // --- stage im2col activation tile with fused ELU while the TDM streams
    for (int e = tid; e < 512; e += 128) {
      int n  = e & 15;
      int kk = e >> 4;
      int kg = kk0 + kk;
      float v = 0.f;
      int t = t0 + n;
      if (kg < K && t < Tout) {
        int dk = kg / CIN, ci = kg & (CIN - 1);      // CIN is a power of two
        int ti = t * STRIDE + dk - PAD;
        if (ti >= 0 && ti < Tin) v = elu_f((float)inb[(size_t)ci * Tin + ti]);
      }
      Bs[kk][n] = (__bf16)v;
    }
    __syncthreads();
    if (mt < MT) {                                   // wave-uniform branch
      __builtin_amdgcn_s_wait_tensorcnt(0);          // weights landed in LDS
      v16bf a, b;
      const int ab = hi ? 8 : 0;                     // A 16x32 bf16 layout
#pragma unroll
      for (int j = 0; j < 16; ++j) {
        int ka = (j < 8) ? (ab + j) : (ab + 8 + j);
        a[j] = As[wave][lm][ka];
      }
      const int kb = hi ? 16 : 0;                    // B 32x16 bf16 layout
#pragma unroll
      for (int j = 0; j < 16; ++j) b[j] = Bs[(j < 8) ? (ab + j) : (ab + 8 + j)][lm];
      acc = __builtin_amdgcn_wmma_f32_16x16x32_bf16(false, a, false, b,
                                                    (short)0, acc, false, false);
    }
    __syncthreads();
  }

  if (mt < MT) {
    int t = t0 + lm;
    if (t < Tout) {
      const int mo = hi ? 8 : 0;                     // C/D: e -> m = e + 8*hi
#pragma unroll
      for (int e = 0; e < 8; ++e) {
        int m = m0 + mo + e;
        float v = acc[e] + bias[m];
        if constexpr (MODE == 0) {
          out_bf[((size_t)bb * cout + m) * Tout + t] = (__bf16)v;
        } else if constexpr (MODE == 1) {
          size_t o = ((size_t)bb * cout + m) * Tout + t;
          out_bf[o] = (__bf16)((float)out_bf[o] + v);  // h = h + conv(...)
        } else {
          out_f32[((size_t)bb * Tout + t) * cout + m] = v;  // r-layout
        }
      }
    }
  }
}

// ---------------- RVQ helpers ----------------------------------------------
__global__ void k_cbnorm(const float* __restrict__ cb, float* __restrict__ cn,
                         int rows) {
  int r = blockIdx.x * blockDim.x + threadIdx.x;
  if (r >= rows) return;
  const float* p = cb + (size_t)r * 128;
  float s = 0.f;
  for (int d = 0; d < 128; ++d) { float v = p[d]; s += v * v; }
  cn[r] = s;
}

__global__ void k_zero(float* __restrict__ p, long n) {
  long i = (long)blockIdx.x * blockDim.x + threadIdx.x;
  if (i < n) p[i] = 0.f;
}

// One block (8 waves) per 16-position tile; 64 code-tiles round-robined over
// waves; full-precision f32 WMMA 16x16x4 for the distance GEMM.
__global__ __launch_bounds__(256)
void k_rvq_argmin(const float* __restrict__ cb, const float* __restrict__ cn,
                  const float* __restrict__ res, int* __restrict__ codes) {
  __shared__ float smin[8][16];
  __shared__ int   sidx[8][16];
  const int tid  = threadIdx.x;
  const int lane = tid & 31;
  const int wave = tid >> 5;
  const int lm   = lane & 15;
  const bool hi  = lane >= 16;
  const int pos0 = blockIdx.x * 16;
  const int ko   = hi ? 2 : 0;             // f32 A/B: lanes>=16 hold K+2

  float bestv = 3.4e38f;
  int   besti = 0;
  for (int mtile = wave; mtile < 64; mtile += 8) {
    v8f acc = {};
    const float* arow = cb  + (size_t)(mtile * 16 + lm) * 128;
    const float* brow = res + (size_t)(pos0 + lm) * 128;
    for (int k0 = 0; k0 < 128; k0 += 4) {
      v2f a = *(const v2f*)(arow + k0 + ko);
      v2f b = *(const v2f*)(brow + k0 + ko);
      acc = __builtin_amdgcn_wmma_f32_16x16x4_f32(false, a, false, b,
                                                  (short)0, acc, false, false);
    }
    const int mo = hi ? 8 : 0;
#pragma unroll
    for (int e = 0; e < 8; ++e) {
      int m = mtile * 16 + mo + e;
      float s = cn[m] - 2.f * acc[e];      // ||r||^2 constant over m: drop
      if (s < bestv) { bestv = s; besti = m; }
    }
  }
  // combine lane L with L^16 (same column n, other half of the m-rows)
  float ov = __shfl_xor(bestv, 16, 32);
  int   oi = __shfl_xor(besti, 16, 32);
  if (ov < bestv || (ov == bestv && oi < besti)) { bestv = ov; besti = oi; }
  if (lane < 16) { smin[wave][lane] = bestv; sidx[wave][lane] = besti; }
  __syncthreads();
  if (wave == 0 && lane < 16) {
    float bv = smin[0][lane]; int bi = sidx[0][lane];
#pragma unroll
    for (int w = 1; w < 8; ++w) {
      float v = smin[w][lane]; int iw = sidx[w][lane];
      if (v < bv || (v == bv && iw < bi)) { bv = v; bi = iw; }
    }
    codes[pos0 + lane] = bi;
  }
}

__global__ void k_rvq_update(const float* __restrict__ cb,
                             const int* __restrict__ codes,
                             float* __restrict__ res, float* __restrict__ quant) {
  int idx = blockIdx.x * blockDim.x + threadIdx.x;
  if (idx >= 12000 * 128) return;
  int pos = idx >> 7, d = idx & 127;
  float s = cb[(size_t)codes[pos] * 128 + d];
  res[idx]   -= s;
  quant[idx] += s;
}

__global__ void k_finalize(const float* __restrict__ quant,
                           const int* __restrict__ codes,
                           float* __restrict__ out) {
  int idx = blockIdx.x * blockDim.x + threadIdx.x;
  const int NF = 16 * 128 * 750;
  if (idx < NF) {                          // feats [B,128,T] from [pos][128]
    int b = idx / (128 * 750);
    int rem = idx % (128 * 750);
    int d = rem / 750, t = rem % 750;
    out[idx] = quant[((size_t)b * 750 + t) * 128 + d];
  } else if (idx < NF + 16 * 8 * 750) {    // codes [B,NQ,T] as float
    int k = idx - NF;
    int b = k / (8 * 750);
    int rem = k % (8 * 750);
    int q = rem / 750, t = rem % 750;
    out[idx] = (float)codes[q * 12000 + b * 750 + t];
  }
}

// ---------------------------------------------------------------------------
template <int CIN, int KSIZE, int STRIDE, int MODE>
static void launch_conv(const __bf16* in, const __bf16* wT, const float* bias,
                        __bf16* obf, float* of32, int cout, int Tin, int Tout,
                        hipStream_t stream) {
  dim3 grid((Tout + 15) / 16, (cout / 16 + 3) / 4, 16);
  k_conv_gemm<CIN, KSIZE, STRIDE, MODE>
      <<<grid, 128, 0, stream>>>(in, wT, bias, obf, of32, cout, Tin, Tout);
}

extern "C" void kernel_launch(void* const* d_in, const int* in_sizes, int n_in,
                              void* d_out, int out_size, void* d_ws, size_t ws_size,
                              hipStream_t stream) {
  (void)in_sizes; (void)n_in; (void)out_size; (void)ws_size;
  const float* x    = (const float*)d_in[0];
  const float* w_in = (const float*)d_in[1];
  const float* b_in = (const float*)d_in[2];
  const float *res_w1[4], *res_b1[4], *res_w2[4], *res_b2[4], *down_w[4], *down_b[4];
  for (int i = 0; i < 4; ++i) {
    res_w1[i] = (const float*)d_in[3 + i];
    res_b1[i] = (const float*)d_in[7 + i];
    res_w2[i] = (const float*)d_in[11 + i];
    res_b2[i] = (const float*)d_in[15 + i];
    down_w[i] = (const float*)d_in[19 + i];
    down_b[i] = (const float*)d_in[23 + i];
  }
  const float* w_out     = (const float*)d_in[27];
  const float* b_out     = (const float*)d_in[28];
  const float* codebooks = (const float*)d_in[29];

  // ---- carve workspace
  char* p = (char*)d_ws;
  auto carve = [&](size_t bytes) -> char* {
    char* q = p; p += (bytes + 255) & ~(size_t)255; return q;
  };
  __bf16* bufA = (__bf16*)carve(122880000ull * sizeof(__bf16));
  __bf16* bufC = (__bf16*)carve(122880000ull * sizeof(__bf16));
  __bf16* bufY = (__bf16*)carve(61440000ull * sizeof(__bf16));
  float*  rbuf = (float*)carve(12000ull * 128 * 4);
  float*  qbuf = (float*)carve(12000ull * 128 * 4);
  float*  cnrm = (float*)carve(8192ull * 4);
  int*    cods = (int*)carve(8ull * 12000 * 4);

  struct ConvDesc { const float* w; const float* b; int cout, cin, k; };
  ConvDesc cd[13];
  const int RAT[4] = {2, 4, 5, 8};
  {
    int c = 32, j = 0;
    for (int i = 0; i < 4; ++i) {
      cd[j++] = { res_w1[i], res_b1[i], c / 2, c, 3 };
      cd[j++] = { res_w2[i], res_b2[i], c, c / 2, 1 };
      cd[j++] = { down_w[i], down_b[i], 2 * c, c, 2 * RAT[i] };
      c *= 2;
    }
    cd[12] = { w_out, b_out, 128, 512, 7 };
  }
  __bf16* wT[13];
  for (int j = 0; j < 13; ++j) {
    size_t n = (size_t)cd[j].cout * cd[j].cin * cd[j].k;
    wT[j] = (__bf16*)carve(n * sizeof(__bf16));
    k_w2bf<<<(int)((n + 255) / 256), 256, 0, stream>>>(cd[j].w, wT[j],
                                                       cd[j].cout, cd[j].cin, cd[j].k);
  }

  // ---- encoder
  {
    long total = 122880000L;
    k_conv_in<<<(int)((total + 255) / 256), 256, 0, stream>>>(x, w_in, b_in, bufA);
  }
  __bf16* A = bufA; __bf16* C = bufC;
  // stage 0: c=32, T=240000, r=2
  launch_conv<32, 3, 1, 0>(A, wT[0], cd[0].b, bufY, nullptr, 16, 240000, 240000, stream);
  launch_conv<16, 1, 1, 1>(bufY, wT[1], cd[1].b, A, nullptr, 32, 240000, 240000, stream);
  launch_conv<32, 4, 2, 0>(A, wT[2], cd[2].b, C, nullptr, 64, 240000, 120000, stream);
  { __bf16* t = A; A = C; C = t; }
  // stage 1: c=64, T=120000, r=4
  launch_conv<64, 3, 1, 0>(A, wT[3], cd[3].b, bufY, nullptr, 32, 120000, 120000, stream);
  launch_conv<32, 1, 1, 1>(bufY, wT[4], cd[4].b, A, nullptr, 64, 120000, 120000, stream);
  launch_conv<64, 8, 4, 0>(A, wT[5], cd[5].b, C, nullptr, 128, 120000, 30000, stream);
  { __bf16* t = A; A = C; C = t; }
  // stage 2: c=128, T=30000, r=5
  launch_conv<128, 3, 1, 0>(A, wT[6], cd[6].b, bufY, nullptr, 64, 30000, 30000, stream);
  launch_conv<64, 1, 1, 1>(bufY, wT[7], cd[7].b, A, nullptr, 128, 30000, 30000, stream);
  launch_conv<128, 10, 5, 0>(A, wT[8], cd[8].b, C, nullptr, 256, 30000, 6000, stream);
  { __bf16* t = A; A = C; C = t; }
  // stage 3: c=256, T=6000, r=8
  launch_conv<256, 3, 1, 0>(A, wT[9], cd[9].b, bufY, nullptr, 128, 6000, 6000, stream);
  launch_conv<128, 1, 1, 1>(bufY, wT[10], cd[10].b, A, nullptr, 256, 6000, 6000, stream);
  launch_conv<256, 16, 8, 0>(A, wT[11], cd[11].b, C, nullptr, 512, 6000, 750, stream);
  { __bf16* t = A; A = C; C = t; }
  // conv_out -> f32 residual buffer in [pos][128] layout
  launch_conv<512, 7, 1, 2>(A, wT[12], cd[12].b, nullptr, rbuf, 128, 750, 750, stream);

  // ---- RVQ (8 stages)
  k_zero<<<(12000 * 128 + 255) / 256, 256, 0, stream>>>(qbuf, 12000L * 128);
  k_cbnorm<<<(8192 + 127) / 128, 128, 0, stream>>>(codebooks, cnrm, 8192);
  for (int q = 0; q < 8; ++q) {
    const float* cbq = codebooks + (size_t)q * 1024 * 128;
    k_rvq_argmin<<<750, 256, 0, stream>>>(cbq, cnrm + q * 1024, rbuf,
                                          cods + q * 12000);
    k_rvq_update<<<(12000 * 128 + 255) / 256, 256, 0, stream>>>(cbq, cods + q * 12000,
                                                                rbuf, qbuf);
  }
  k_finalize<<<(1632000 + 255) / 256, 256, 0, stream>>>(qbuf, cods, (float*)d_out);
}
